// CrossLayerSparseMoE_63067299775269
// MI455X (gfx1250) — matrix operand order repaired and checked by reading, not compiled, not run on hardware
//
#include <hip/hip_runtime.h>

// ---------------- problem constants ----------------
#define TOK   8192      // B*S
#define DIM   512       // D
#define NEXP  8         // E
#define KTOP  2         // K
#define HID   2048      // H = 4*D
#define CAPMAX 2048     // TOK*KTOP/NEXP (upper bound on capacity)

#define LDSB_STRIDE 40  // 32 bf16 payload + 8 pad (80B rows -> bank spread)

typedef __attribute__((ext_vector_type(16))) __bf16   v16bf;
typedef __attribute__((ext_vector_type(8)))  float    v8f;
typedef __attribute__((ext_vector_type(4)))  unsigned int v4u;
typedef __attribute__((ext_vector_type(4)))  int       v4i;

union Frag { v16bf v; v4u q[2]; };

#if __has_builtin(__builtin_amdgcn_global_load_async_to_lds_b128) && \
    __has_builtin(__builtin_amdgcn_s_wait_asynccnt)
#define USE_ASYNC 1
#define GPTR(p) ((__attribute__((address_space(1))) v4i*)((v4i*)(p)))
#define LPTR(p) ((__attribute__((address_space(3))) v4i*)((v4i*)(p)))
#else
#define USE_ASYNC 0
#endif

__device__ __forceinline__ unsigned short f2bf(float f) {
    unsigned int u = __float_as_uint(f);
    u += 0x7FFFu + ((u >> 16) & 1u);   // round-to-nearest-even
    return (unsigned short)(u >> 16);
}

// ---------------- bf16 conversion / transpose kernels ----------------
__global__ void k_conv_x(const float* __restrict__ x, unsigned short* __restrict__ xb) {
    int i = blockIdx.x * 256 + threadIdx.x;
    xb[i] = f2bf(x[i]);
}

// W1: [E][D][H] fp32 -> W1t: [E][H][D] bf16  (N-major so B rows are contiguous in K)
__global__ void k_conv_w1(const float* __restrict__ W1, unsigned short* __restrict__ W1t) {
    int i = blockIdx.x * 256 + threadIdx.x;        // over E*H*D
    int e = i / (HID * DIM);
    int r = i - e * (HID * DIM);
    int n = r / DIM;
    int k = r - n * DIM;
    W1t[i] = f2bf(W1[(size_t)e * DIM * HID + (size_t)k * HID + n]);
}

// W2: [E][H][D] fp32 -> W2t: [E][D][H] bf16
__global__ void k_conv_w2(const float* __restrict__ W2, unsigned short* __restrict__ W2t) {
    int i = blockIdx.x * 256 + threadIdx.x;        // over E*D*H
    int e = i / (DIM * HID);
    int r = i - e * (DIM * HID);
    int d = r / HID;
    int h = r - d * HID;
    W2t[i] = f2bf(W2[(size_t)e * HID * DIM + (size_t)h * DIM + d]);
}

// ---------------- router: one wave32 per token ----------------
__global__ __launch_bounds__(256) void k_router(
        const float* __restrict__ x, const float* __restrict__ noise,
        const float* __restrict__ W_r, const float* __restrict__ b_r,
        const float* __restrict__ W_n, const float* __restrict__ b_n,
        const float* __restrict__ W_s, const float* __restrict__ b_s,
        int* __restrict__ topE, float* __restrict__ topG, int* __restrict__ skipA) {
    int wave = threadIdx.x >> 5;
    int lane = threadIdx.x & 31;
    int t = blockIdx.x * 8 + wave;

    const float* xr = x + (size_t)t * DIM + lane * 16;
    float aR[NEXP] = {}, aN[NEXP] = {};
    float aS = 0.f;
    #pragma unroll
    for (int j = 0; j < 16; ++j) {
        int d = lane * 16 + j;
        float xv = xr[j];
        const float* wr = W_r + d * NEXP;
        const float* wn = W_n + d * NEXP;
        #pragma unroll
        for (int e = 0; e < NEXP; ++e) {
            aR[e] = fmaf(xv, wr[e], aR[e]);
            aN[e] = fmaf(xv, wn[e], aN[e]);
        }
        aS = fmaf(xv, W_s[d], aS);
    }
    #pragma unroll
    for (int off = 16; off; off >>= 1) {
        #pragma unroll
        for (int e = 0; e < NEXP; ++e) {
            aR[e] += __shfl_xor(aR[e], off, 32);
            aN[e] += __shfl_xor(aN[e], off, 32);
        }
        aS += __shfl_xor(aS, off, 32);
    }

    if (lane == 0) {
        float noisy[NEXP];
        #pragma unroll
        for (int e = 0; e < NEXP; ++e) {
            float nl = aN[e] + b_n[e];
            float sp = (nl > 0.f) ? (nl + log1pf(expf(-nl))) : log1pf(expf(nl));
            noisy[e] = aR[e] + b_r[e] + noise[(size_t)t * NEXP + e] * sp;
        }
        int i0 = 0;
        #pragma unroll
        for (int e = 1; e < NEXP; ++e) if (noisy[e] > noisy[i0]) i0 = e;
        int i1 = (i0 == 0) ? 1 : 0;
        #pragma unroll
        for (int e = 0; e < NEXP; ++e)
            if (e != i0 && noisy[e] > noisy[i1]) i1 = e;
        float ex = expf(noisy[i1] - noisy[i0]);
        float g0 = 1.f / (1.f + ex);
        float g1 = ex * g0;
        topE[2 * t]     = i0;
        topE[2 * t + 1] = i1;
        topG[2 * t]     = g0;
        topG[2 * t + 1] = g1;
        float sprob = 1.f / (1.f + expf(-(aS + b_s[0])));
        skipA[t] = (sprob > 0.5f) ? 1 : 0;
    }
}

// ---------------- capacity scan: single workgroup, order-preserving ----------------
__global__ __launch_bounds__(1024) void k_scan(
        const int* __restrict__ topE, const float* __restrict__ topG,
        const int* __restrict__ skipA,
        int* __restrict__ gather, float* __restrict__ gateL, int* __restrict__ cnt) {
    __shared__ int s[1024];
    __shared__ int capS;
    int tid = threadIdx.x;
    int t0 = tid * 8;               // contiguous chunk keeps flat token order

    int ns = 0;
    #pragma unroll
    for (int j = 0; j < 8; ++j) ns += (skipA[t0 + j] == 0);
    s[tid] = ns; __syncthreads();
    for (int off = 512; off > 0; off >>= 1) {
        if (tid < off) s[tid] += s[tid + off];
        __syncthreads();
    }
    if (tid == 0) capS = (s[0] * KTOP) / NEXP;
    __syncthreads();
    int cap = capS;

    for (int e = 0; e < NEXP; ++e) {
        int match[8]; float gv[8];
        int c = 0;
        #pragma unroll
        for (int j = 0; j < 8; ++j) {
            int t = t0 + j;
            int m = 0; float g = 0.f;
            if (!skipA[t]) {
                if (topE[2 * t] == e)           { m = 1; g = topG[2 * t]; }
                else if (topE[2 * t + 1] == e)  { m = 1; g = topG[2 * t + 1]; }
            }
            match[j] = m; gv[j] = g; c += m;
        }
        s[tid] = c; __syncthreads();
        for (int off = 1; off < 1024; off <<= 1) {
            int tmp = (tid >= off) ? s[tid - off] : 0;
            __syncthreads();
            s[tid] += tmp;
            __syncthreads();
        }
        int base  = s[tid] - c;
        int total = s[1023];
        __syncthreads();
        int r = base;
        #pragma unroll
        for (int j = 0; j < 8; ++j) {
            if (match[j]) {
                if (r < cap) {
                    gather[e * CAPMAX + r] = t0 + j;
                    gateL[e * CAPMAX + r]  = gv[j];
                }
                ++r;
            }
        }
        if (tid == 0) cnt[e] = (total < cap) ? total : cap;
        __syncthreads();
    }
}

// ---------------- out init: skip => identity, else 0 ----------------
__global__ void k_init_out(const float* __restrict__ x, const int* __restrict__ skipA,
                           float* __restrict__ out) {
    int i = blockIdx.x * 256 + threadIdx.x;
    int t = i / DIM;
    out[i] = skipA[t] ? x[i] : 0.f;
}

// ---------------- GEMM1: h = relu(Xg @ W1 + b1), bf16 WMMA, LDS-staged B ----------
// grid: (HID/64, CAPMAX/256, NEXP), block 256.
// Block tile 256(M) x 64(N): 8 waves split M; the 64x32 B tile per K-step is staged
// once into LDS (async DMA, double-buffered, constant buffer indices) and shared
// by all 8 waves. A fragments are register double-buffered via two NAMED buffers
// (no dynamic indexing -> no scratch spill).
__global__ __launch_bounds__(256) void k_gemm1(
        const unsigned short* __restrict__ xb,
        const unsigned short* __restrict__ W1t, const float* __restrict__ b1,
        const int* __restrict__ gather, const int* __restrict__ cnt,
        unsigned short* __restrict__ hbuf) {
    int e = blockIdx.z;
    int count = cnt[e];
    if (blockIdx.y * 256 >= count) return;          // uniform exit

    __shared__ unsigned short smemB[2][64 * LDSB_STRIDE];

    int wave = threadIdx.x >> 5, lane = threadIdx.x & 31;
    int half = lane >> 4, row = lane & 15;
    int nBase = blockIdx.x * 64;
    int mBase = blockIdx.y * 256 + wave * 32;

    int rowG0 = mBase + row;
    int rowG1 = rowG0 + 16;
    int tok0 = (rowG0 < count) ? gather[e * CAPMAX + rowG0] : 0;
    int tok1 = (rowG1 < count) ? gather[e * CAPMAX + rowG1] : 0;
    const unsigned short* ap0 = xb + (size_t)tok0 * DIM + half * 8;
    const unsigned short* ap1 = xb + (size_t)tok1 * DIM + half * 8;
    const unsigned short* bbase = W1t + (size_t)e * HID * DIM;

    // staging task: 256 threads cover 64 rows x 4 x 16B chunks of the B tile
    int sr = threadIdx.x >> 2;        // 0..63  (B row within tile)
    int sc = threadIdx.x & 3;         // 16B chunk
    const unsigned short* sgp = bbase + (size_t)(nBase + sr) * DIM + sc * 8;
    unsigned short* slp0 = &smemB[0][sr * LDSB_STRIDE + sc * 8];
    unsigned short* slp1 = &smemB[1][sr * LDSB_STRIDE + sc * 8];

    Frag A0[2], A1[2];
    v8f acc[2][4] = {};

    auto loadA = [&](Frag (&A)[2], int k0) {
        A[0].q[0] = *(const v4u*)(ap0 + k0);
        A[0].q[1] = *(const v4u*)(ap0 + k0 + 16);
        A[1].q[0] = *(const v4u*)(ap1 + k0);
        A[1].q[1] = *(const v4u*)(ap1 + k0 + 16);
    };
    auto stageTo = [&](unsigned short* lp, int k0) {
        const unsigned short* gp = sgp + k0;
#if USE_ASYNC
        __builtin_amdgcn_global_load_async_to_lds_b128(GPTR(gp), LPTR(lp), 0, 0);
#else
        *(v4u*)lp = *(const v4u*)gp;
#endif
    };
    // one 32-K step: consume (Acur, LDS bufCur), prefetch (Anext, other LDS buf)
    auto step = [&](Frag (&Acur)[2], Frag (&Anext)[2],
                    const unsigned short* lbCur, unsigned short* lpNext, int kNext) {
        __syncthreads();                          // all waves done reading other buf
        if (kNext < DIM) {
            stageTo(lpNext, kNext);
#if USE_ASYNC
            __builtin_amdgcn_s_wait_asynccnt(1);  // current buf's DMA done
#endif
            loadA(Anext, kNext);
        } else {
#if USE_ASYNC
            __builtin_amdgcn_s_wait_asynccnt(0);
#endif
        }
        __syncthreads();                          // current buf visible block-wide
        #pragma unroll
        for (int nt = 0; nt < 4; ++nt) {
            Frag Bf;
            const unsigned short* bp = lbCur + (nt * 16 + row) * LDSB_STRIDE + half * 8;
            Bf.q[0] = *(const v4u*)bp;
            Bf.q[1] = *(const v4u*)(bp + 16);
            acc[0][nt] = __builtin_amdgcn_wmma_f32_16x16x32_bf16(
                false, Acur[0].v, false, Bf.v, (short)0, acc[0][nt], false, false);
            acc[1][nt] = __builtin_amdgcn_wmma_f32_16x16x32_bf16(
                false, Acur[1].v, false, Bf.v, (short)0, acc[1][nt], false, false);
        }
    };

    stageTo(slp0, 0);
    loadA(A0, 0);
    for (int k0 = 0; k0 < DIM; k0 += 64) {
        step(A0, A1, &smemB[0][0], slp1, k0 + 32);
        step(A1, A0, &smemB[1][0], slp0, k0 + 64);
    }

    unsigned short* hb = hbuf + (size_t)e * CAPMAX * HID;
    #pragma unroll
    for (int nt = 0; nt < 4; ++nt) {
        int n = nBase + nt * 16 + row;
        float bias = b1[e * HID + n];
        #pragma unroll
        for (int mt = 0; mt < 2; ++mt) {
            #pragma unroll
            for (int r = 0; r < 8; ++r) {
                int m = mBase + mt * 16 + half * 8 + r;
                float v = acc[mt][nt][r] + bias;
                v = fmaxf(v, 0.f);
                hb[(size_t)m * HID + n] = f2bf(v);
            }
        }
    }
}

// ---------------- GEMM2: y = h @ W2 + b2; out[tok] += gate*y ----------------
// grid: (DIM/64, CAPMAX/256, NEXP), block 256. Same LDS-staged-B scheme, K=HID.
__global__ __launch_bounds__(256) void k_gemm2(
        const unsigned short* __restrict__ hbuf,
        const unsigned short* __restrict__ W2t, const float* __restrict__ b2,
        const int* __restrict__ gather, const float* __restrict__ gateL,
        const int* __restrict__ cnt, float* __restrict__ out) {
    int e = blockIdx.z;
    int count = cnt[e];
    if (blockIdx.y * 256 >= count) return;

    __shared__ unsigned short smemB[2][64 * LDSB_STRIDE];

    int wave = threadIdx.x >> 5, lane = threadIdx.x & 31;
    int half = lane >> 4, row = lane & 15;
    int nBase = blockIdx.x * 64;
    int mBase = blockIdx.y * 256 + wave * 32;

    int rowA0 = mBase + row;          // < CAPMAX
    int rowA1 = rowA0 + 16;
    const unsigned short* ap0 = hbuf + ((size_t)e * CAPMAX + rowA0) * HID + half * 8;
    const unsigned short* ap1 = hbuf + ((size_t)e * CAPMAX + rowA1) * HID + half * 8;
    const unsigned short* bbase = W2t + (size_t)e * DIM * HID;

    int sr = threadIdx.x >> 2;
    int sc = threadIdx.x & 3;
    const unsigned short* sgp = bbase + (size_t)(nBase + sr) * HID + sc * 8;
    unsigned short* slp0 = &smemB[0][sr * LDSB_STRIDE + sc * 8];
    unsigned short* slp1 = &smemB[1][sr * LDSB_STRIDE + sc * 8];

    Frag A0[2], A1[2];
    v8f acc[2][4] = {};

    auto loadA = [&](Frag (&A)[2], int k0) {
        A[0].q[0] = *(const v4u*)(ap0 + k0);
        A[0].q[1] = *(const v4u*)(ap0 + k0 + 16);
        A[1].q[0] = *(const v4u*)(ap1 + k0);
        A[1].q[1] = *(const v4u*)(ap1 + k0 + 16);
    };
    auto stageTo = [&](unsigned short* lp, int k0) {
        const unsigned short* gp = sgp + k0;
#if USE_ASYNC
        __builtin_amdgcn_global_load_async_to_lds_b128(GPTR(gp), LPTR(lp), 0, 0);
#else
        *(v4u*)lp = *(const v4u*)gp;
#endif
    };
    auto step = [&](Frag (&Acur)[2], Frag (&Anext)[2],
                    const unsigned short* lbCur, unsigned short* lpNext, int kNext) {
        __syncthreads();
        if (kNext < HID) {
            stageTo(lpNext, kNext);
#if USE_ASYNC
            __builtin_amdgcn_s_wait_asynccnt(1);
#endif
            loadA(Anext, kNext);
        } else {
#if USE_ASYNC
            __builtin_amdgcn_s_wait_asynccnt(0);
#endif
        }
        __syncthreads();
        #pragma unroll
        for (int nt = 0; nt < 4; ++nt) {
            Frag Bf;
            const unsigned short* bp = lbCur + (nt * 16 + row) * LDSB_STRIDE + half * 8;
            Bf.q[0] = *(const v4u*)bp;
            Bf.q[1] = *(const v4u*)(bp + 16);
            acc[0][nt] = __builtin_amdgcn_wmma_f32_16x16x32_bf16(
                false, Acur[0].v, false, Bf.v, (short)0, acc[0][nt], false, false);
            acc[1][nt] = __builtin_amdgcn_wmma_f32_16x16x32_bf16(
                false, Acur[1].v, false, Bf.v, (short)0, acc[1][nt], false, false);
        }
    };

    stageTo(slp0, 0);
    loadA(A0, 0);
    for (int k0 = 0; k0 < HID; k0 += 64) {
        step(A0, A1, &smemB[0][0], slp1, k0 + 32);
        step(A1, A0, &smemB[1][0], slp0, k0 + 64);
    }

    #pragma unroll
    for (int mt = 0; mt < 2; ++mt) {
        #pragma unroll
        for (int r = 0; r < 8; ++r) {
            int m = mBase + mt * 16 + half * 8 + r;
            if (m < count) {
                int tok  = gather[e * CAPMAX + m];
                float w  = gateL[e * CAPMAX + m];
                #pragma unroll
                for (int nt = 0; nt < 4; ++nt) {
                    int n = nBase + nt * 16 + row;
                    float v = acc[mt][nt][r] + b2[e * DIM + n];
                    atomicAdd(&out[(size_t)tok * DIM + n], v * w);
                }
            }
        }
    }
}

// ---------------- host launcher ----------------
extern "C" void kernel_launch(void* const* d_in, const int* in_sizes, int n_in,
                              void* d_out, int out_size, void* d_ws, size_t ws_size,
                              hipStream_t stream) {
    (void)in_sizes; (void)n_in; (void)out_size; (void)ws_size;
    const float* x    = (const float*)d_in[0];
    const float* noise= (const float*)d_in[1];
    const float* W_r  = (const float*)d_in[2];
    const float* b_r  = (const float*)d_in[3];
    const float* W_n  = (const float*)d_in[4];
    const float* b_n  = (const float*)d_in[5];
    const float* W_s  = (const float*)d_in[6];
    const float* b_s  = (const float*)d_in[7];
    const float* W1   = (const float*)d_in[8];
    const float* b1   = (const float*)d_in[9];
    const float* W2   = (const float*)d_in[10];
    const float* b2   = (const float*)d_in[11];
    float* out = (float*)d_out;

    char* ws = (char*)d_ws;
    size_t off = 0;
    auto take = [&](size_t bytes) -> char* {
        char* p = ws + off;
        off = (off + bytes + 255) & ~(size_t)255;
        return p;
    };
    unsigned short* xb    = (unsigned short*)take((size_t)TOK * DIM * 2);
    unsigned short* W1t   = (unsigned short*)take((size_t)NEXP * HID * DIM * 2);
    unsigned short* W2t   = (unsigned short*)take((size_t)NEXP * DIM * HID * 2);
    int*            topE  = (int*)take((size_t)TOK * 2 * 4);
    float*          topG  = (float*)take((size_t)TOK * 2 * 4);
    int*            skipA = (int*)take((size_t)TOK * 4);
    int*            gat   = (int*)take((size_t)NEXP * CAPMAX * 4);
    float*          gteL  = (float*)take((size_t)NEXP * CAPMAX * 4);
    int*            cnt   = (int*)take(64);
    unsigned short* hbuf  = (unsigned short*)take((size_t)NEXP * CAPMAX * HID * 2);

    k_conv_x <<<TOK * DIM / 256, 256, 0, stream>>>(x, xb);
    k_conv_w1<<<NEXP * HID * DIM / 256, 256, 0, stream>>>(W1, W1t);
    k_conv_w2<<<NEXP * DIM * HID / 256, 256, 0, stream>>>(W2, W2t);
    k_router <<<TOK / 8, 256, 0, stream>>>(x, noise, W_r, b_r, W_n, b_n, W_s, b_s,
                                           topE, topG, skipA);
    k_scan   <<<1, 1024, 0, stream>>>(topE, topG, skipA, gat, gteL, cnt);
    k_init_out<<<TOK * DIM / 256, 256, 0, stream>>>(x, skipA, out);
    k_gemm1  <<<dim3(HID / 64, CAPMAX / 256, NEXP), 256, 0, stream>>>(
                  xb, W1t, b1, gat, cnt, hbuf);
    k_gemm2  <<<dim3(DIM / 64, CAPMAX / 256, NEXP), 256, 0, stream>>>(
                  hbuf, W2t, b2, gat, gteL, cnt, out);
}
